// AttentionBranch_64707977282009
// MI455X (gfx1250) — matrix operand order repaired and checked by reading, not compiled, not run on hardware
//
#include <hip/hip_runtime.h>

typedef __attribute__((ext_vector_type(16))) __bf16 v16bf;
typedef __attribute__((ext_vector_type(8)))  __bf16 v8bf;
typedef __attribute__((ext_vector_type(8)))  float  v8f;

constexpr int CH   = 512;     // channels
constexpr int HW   = 80;      // pooled spatial side
constexpr int SP   = 6400;    // 80*80
constexpr int PADW = 82;      // padded side

__device__ __forceinline__ unsigned short f32_bf16(float f) {
    unsigned int u = __float_as_uint(f);
    u += 0x7FFFu + ((u >> 16) & 1u);      // round-to-nearest-even
    return (unsigned short)(u >> 16);
}

union V16U { v16bf v; v8bf h[2]; };

// A-matrix 16x32 bf16, k order chosen here (B matches): lane half h holds K = h*16..h*16+15 contiguous.
// `p` must already include the +h*16 element offset.
__device__ __forceinline__ v16bf load_a(const unsigned short* p) {
    V16U t;
    t.h[0] = *(const v8bf*)(p);
    t.h[1] = *(const v8bf*)(p + 8);
    return t.v;
}
// B-matrix 32x16 bf16, k order chosen here (A matches): lane half h holds K = h*16..h*16+15 contiguous.
// `p` must already include the +h*16 element offset.
__device__ __forceinline__ v16bf load_b(const unsigned short* p) {
    V16U t;
    t.h[0] = *(const v8bf*)(p);
    t.h[1] = *(const v8bf*)(p + 8);
    return t.v;
}

__global__ void zero_u32(unsigned int* __restrict__ p, int n) {
    int i = blockIdx.x * 256 + threadIdx.x;
    if (i < n) p[i] = 0u;
}

// w f32 [O][I][3][3] -> wt bf16 [tap][O][I]
__global__ void __launch_bounds__(256)
cvt_w(const float* __restrict__ w, unsigned short* __restrict__ wt) {
    int tid = blockIdx.x * 256 + threadIdx.x;           // 9*512*512
    int i   = tid & (CH - 1);
    int o   = (tid >> 9) & (CH - 1);
    int tap = tid >> 18;
    wt[tid] = f32_bf16(w[((size_t)o * CH + i) * 9 + tap]);
}

// x f32 NCHW [512][160][160] -> pad0 bf16 NHWC-padded [82][82][512] (interior)
__global__ void __launch_bounds__(256)
pool_kernel(const float* __restrict__ x, unsigned short* __restrict__ pad0) {
    int tid = blockIdx.x * 256 + threadIdx.x;           // 512*6400
    int s = tid % SP;
    int c = tid / SP;
    int y = s / HW, xx = s % HW;
    const float* p = x + (size_t)c * (160 * 160) + (2 * y) * 160 + 2 * xx;
    float v = fmaxf(fmaxf(p[0], p[1]), fmaxf(p[160], p[161]));
    pad0[((size_t)((y + 1) * PADW + (xx + 1))) * CH + c] = f32_bf16(v);
}

// Implicit-GEMM conv3x3: out[s][o] = sum_{tap,c} wt[tap][o][c] * inPad[s+shift][c]
// Wave tile 64(M=o) x 32(N=s): 4x2 WMMA accumulators; block = 8 waves (2M x 4N) = 128x128
__global__ void __launch_bounds__(256)
conv_gemm(const unsigned short* __restrict__ inPad,   // bf16 [82][82][512]
          const unsigned short* __restrict__ wt,      // bf16 [9][512][512]
          float* __restrict__ out)                    // f32  [6400][512]
{
    const int lane = threadIdx.x & 31;
    const int wave = threadIdx.x >> 5;
    const int wm = wave >> 2;            // 0..1
    const int wn = wave & 3;             // 0..3
    const int m_base = blockIdx.y * 128 + wm * 64;    // out-channel base (64 wide)
    const int n_base = blockIdx.x * 128 + wn * 32;    // spatial base (32 wide)
    const int l16 = lane & 15;
    const int half = lane >> 4;
    const int ka = half * 16;    // A operand K offset
    const int kb = half * 16;    // B operand K offset

    int s0 = n_base + l16, s1 = s0 + 16;
    int y0 = s0 / HW, x0 = s0 % HW;
    int y1 = s1 / HW, x1 = s1 % HW;

    v8f acc[4][2];
    #pragma unroll
    for (int mi = 0; mi < 4; ++mi)
        #pragma unroll
        for (int ni = 0; ni < 2; ++ni) acc[mi][ni] = (v8f){};

    for (int tap = 0; tap < 9; ++tap) {
        const int ky = tap / 3, kx = tap % 3;
        const unsigned short* arow = wt + (size_t)tap * CH * CH + (m_base + l16) * CH + ka;
        const unsigned short* brow0 = inPad + ((size_t)((y0 + ky) * PADW + (x0 + kx))) * CH + kb;
        const unsigned short* brow1 = inPad + ((size_t)((y1 + ky) * PADW + (x1 + kx))) * CH + kb;
        for (int kc = 0; kc < CH; kc += 32) {
            __builtin_prefetch((const void*)(brow0 + kc + 64), 0, 3);   // WGP-scope prefetch
            v16bf b0 = load_b(brow0 + kc);
            v16bf b1 = load_b(brow1 + kc);
            v16bf a0 = load_a(arow + kc);
            v16bf a1 = load_a(arow + kc + 16 * CH);
            v16bf a2 = load_a(arow + kc + 32 * CH);
            v16bf a3 = load_a(arow + kc + 48 * CH);
            acc[0][0] = __builtin_amdgcn_wmma_f32_16x16x32_bf16(false, a0, false, b0, (short)0, acc[0][0], false, false);
            acc[0][1] = __builtin_amdgcn_wmma_f32_16x16x32_bf16(false, a0, false, b1, (short)0, acc[0][1], false, false);
            acc[1][0] = __builtin_amdgcn_wmma_f32_16x16x32_bf16(false, a1, false, b0, (short)0, acc[1][0], false, false);
            acc[1][1] = __builtin_amdgcn_wmma_f32_16x16x32_bf16(false, a1, false, b1, (short)0, acc[1][1], false, false);
            acc[2][0] = __builtin_amdgcn_wmma_f32_16x16x32_bf16(false, a2, false, b0, (short)0, acc[2][0], false, false);
            acc[2][1] = __builtin_amdgcn_wmma_f32_16x16x32_bf16(false, a2, false, b1, (short)0, acc[2][1], false, false);
            acc[3][0] = __builtin_amdgcn_wmma_f32_16x16x32_bf16(false, a3, false, b0, (short)0, acc[3][0], false, false);
            acc[3][1] = __builtin_amdgcn_wmma_f32_16x16x32_bf16(false, a3, false, b1, (short)0, acc[3][1], false, false);
        }
    }
    // D tile: VGPR r -> M = r + 8*half, N = l16  => 8 contiguous channels per store
    #pragma unroll
    for (int ni = 0; ni < 2; ++ni) {
        float* p = out + (size_t)(n_base + ni * 16 + l16) * CH + m_base + half * 8;
        #pragma unroll
        for (int mi = 0; mi < 4; ++mi)
            *(v8f*)(p + mi * 16) = acc[mi][ni];
    }
}

// Per-channel batch mean / inv-std (biased variance, eps=1e-5)
__global__ void __launch_bounds__(256)
bn_stats(const float* __restrict__ conv, float* __restrict__ mean, float* __restrict__ istd) {
    __shared__ float s1[256], s2[256];
    const int c = blockIdx.x;
    float sum = 0.f, sq = 0.f;
    for (int s = threadIdx.x; s < SP; s += 256) {
        float v = conv[(size_t)s * CH + c];
        sum += v; sq += v * v;
    }
    s1[threadIdx.x] = sum; s2[threadIdx.x] = sq;
    __syncthreads();
    for (int off = 128; off > 0; off >>= 1) {
        if (threadIdx.x < off) {
            s1[threadIdx.x] += s1[threadIdx.x + off];
            s2[threadIdx.x] += s2[threadIdx.x + off];
        }
        __syncthreads();
    }
    if (threadIdx.x == 0) {
        float m = s1[0] / (float)SP;
        float v = s2[0] / (float)SP - m * m;
        mean[c] = m;
        istd[c] = rsqrtf(v + 1e-5f);
    }
}

// BN + ReLU + f32->bf16; dst either padded NHWC (next conv) or flat NHWC (gram)
__global__ void __launch_bounds__(256)
bn_apply(const float* __restrict__ conv, const float* __restrict__ mean,
         const float* __restrict__ istd, const float* __restrict__ gamma,
         const float* __restrict__ beta, unsigned short* __restrict__ dst, int padded) {
    int tid = blockIdx.x * 256 + threadIdx.x;           // SP*CH
    int c = tid & (CH - 1);
    int s = tid >> 9;
    float v = conv[tid];
    float y = fmaxf((v - mean[c]) * istd[c] * gamma[c] + beta[c], 0.f);
    size_t idx;
    if (padded) {
        int yy = s / HW + 1, xx = s % HW + 1;
        idx = ((size_t)(yy * PADW + xx)) * CH + c;
    } else {
        idx = (size_t)tid;
    }
    dst[idx] = f32_bf16(y);
}

// scores[i][j] = sum_c f[i][c] * f[j][c]   (f bf16 [6400][512])
// Wave tile 64(i) x 32(j); block = 8 waves (2x4) = 128x128
__global__ void __launch_bounds__(256)
gram_gemm(const unsigned short* __restrict__ f, float* __restrict__ out) {
    const int lane = threadIdx.x & 31;
    const int wave = threadIdx.x >> 5;
    const int wm = wave >> 2;
    const int wn = wave & 3;
    const int i_base = blockIdx.y * 128 + wm * 64;
    const int j_base = blockIdx.x * 128 + wn * 32;
    const int l16 = lane & 15;
    const int half = lane >> 4;

    const unsigned short* arow  = f + (size_t)(i_base + l16) * CH + half * 16;
    const unsigned short* brow0 = f + (size_t)(j_base + l16) * CH + half * 16;
    const unsigned short* brow1 = brow0 + 16 * CH;

    v8f acc[4][2];
    #pragma unroll
    for (int mi = 0; mi < 4; ++mi)
        #pragma unroll
        for (int ni = 0; ni < 2; ++ni) acc[mi][ni] = (v8f){};

    for (int kc = 0; kc < CH; kc += 32) {
        __builtin_prefetch((const void*)(arow + kc + 64), 0, 3);        // WGP-scope prefetch
        v16bf b0 = load_b(brow0 + kc);
        v16bf b1 = load_b(brow1 + kc);
        v16bf a0 = load_a(arow + kc);
        v16bf a1 = load_a(arow + kc + 16 * CH);
        v16bf a2 = load_a(arow + kc + 32 * CH);
        v16bf a3 = load_a(arow + kc + 48 * CH);
        acc[0][0] = __builtin_amdgcn_wmma_f32_16x16x32_bf16(false, a0, false, b0, (short)0, acc[0][0], false, false);
        acc[0][1] = __builtin_amdgcn_wmma_f32_16x16x32_bf16(false, a0, false, b1, (short)0, acc[0][1], false, false);
        acc[1][0] = __builtin_amdgcn_wmma_f32_16x16x32_bf16(false, a1, false, b0, (short)0, acc[1][0], false, false);
        acc[1][1] = __builtin_amdgcn_wmma_f32_16x16x32_bf16(false, a1, false, b1, (short)0, acc[1][1], false, false);
        acc[2][0] = __builtin_amdgcn_wmma_f32_16x16x32_bf16(false, a2, false, b0, (short)0, acc[2][0], false, false);
        acc[2][1] = __builtin_amdgcn_wmma_f32_16x16x32_bf16(false, a2, false, b1, (short)0, acc[2][1], false, false);
        acc[3][0] = __builtin_amdgcn_wmma_f32_16x16x32_bf16(false, a3, false, b0, (short)0, acc[3][0], false, false);
        acc[3][1] = __builtin_amdgcn_wmma_f32_16x16x32_bf16(false, a3, false, b1, (short)0, acc[3][1], false, false);
    }
    #pragma unroll
    for (int mi = 0; mi < 4; ++mi) {
        #pragma unroll
        for (int r = 0; r < 8; ++r) {
            int i0 = i_base + mi * 16 + r + 8 * half;
            out[(size_t)i0 * SP + j_base + l16]      = acc[mi][0][r];
            out[(size_t)i0 * SP + j_base + 16 + l16] = acc[mi][1][r];
        }
    }
}

// In-place row softmax over [6400][6400]; 25 values/thread held in registers
__global__ void __launch_bounds__(256)
softmax_rows(float* __restrict__ a) {
    __shared__ float red[256];
    float* p = a + (size_t)blockIdx.x * SP;
    float v[25];
    float m = -3.402823e38f;
    for (int i = 0; i < 25; ++i) {
        v[i] = p[threadIdx.x + i * 256];
        m = fmaxf(m, v[i]);
    }
    red[threadIdx.x] = m; __syncthreads();
    for (int off = 128; off > 0; off >>= 1) {
        if (threadIdx.x < off) red[threadIdx.x] = fmaxf(red[threadIdx.x], red[threadIdx.x + off]);
        __syncthreads();
    }
    m = red[0]; __syncthreads();
    float sum = 0.f;
    for (int i = 0; i < 25; ++i) { v[i] = __expf(v[i] - m); sum += v[i]; }
    red[threadIdx.x] = sum; __syncthreads();
    for (int off = 128; off > 0; off >>= 1) {
        if (threadIdx.x < off) red[threadIdx.x] += red[threadIdx.x + off];
        __syncthreads();
    }
    float inv = 1.f / red[0];
    for (int i = 0; i < 25; ++i) p[threadIdx.x + i * 256] = v[i] * inv;
}

extern "C" void kernel_launch(void* const* d_in, const int* in_sizes, int n_in,
                              void* d_out, int out_size, void* d_ws, size_t ws_size,
                              hipStream_t stream) {
    const float* x  = (const float*)d_in[0];
    // d_in[1] = upsampling_size (unused: upsample not part of output)
    const float* w1 = (const float*)d_in[2];
    // conv biases d_in[3]/d_in[7] cancel exactly under training-mode BN -> skipped
    const float* g1 = (const float*)d_in[4];
    const float* b1 = (const float*)d_in[5];
    const float* w2 = (const float*)d_in[6];
    const float* g2 = (const float*)d_in[8];
    const float* b2 = (const float*)d_in[9];
    float* out = (float*)d_out;

    char* ws = (char*)d_ws;
    unsigned short* wt1  = (unsigned short*)(ws + 0);          //  4,718,592 B
    unsigned short* wt2  = (unsigned short*)(ws + 4718592);    //  4,718,592 B
    unsigned short* pad0 = (unsigned short*)(ws + 9437184);    //  6,885,376 B
    unsigned short* pad1 = (unsigned short*)(ws + 16322560);   //  6,885,376 B
    float*          cvt  = (float*)         (ws + 23207936);   // 13,107,200 B
    unsigned short* h2   = (unsigned short*)(ws + 36315136);   //  6,553,600 B
    float*          mb   = (float*)         (ws + 42868736);   //  2,048 B
    float*          ib   = (float*)         (ws + 42870784);   //  2,048 B

    // zero both padded activation buffers (contiguous): 2*82*82*512 bf16 = 3,442,688 u32
    zero_u32<<<(3442688 + 255) / 256, 256, 0, stream>>>((unsigned int*)pad0, 3442688);
    cvt_w<<<9216, 256, 0, stream>>>(w1, wt1);
    cvt_w<<<9216, 256, 0, stream>>>(w2, wt2);
    pool_kernel<<<(CH * SP) / 256, 256, 0, stream>>>(x, pad0);

    conv_gemm<<<dim3(50, 4), 256, 0, stream>>>(pad0, wt1, cvt);
    bn_stats<<<CH, 256, 0, stream>>>(cvt, mb, ib);
    bn_apply<<<(CH * SP) / 256, 256, 0, stream>>>(cvt, mb, ib, g1, b1, pad1, 1);

    conv_gemm<<<dim3(50, 4), 256, 0, stream>>>(pad1, wt2, cvt);
    bn_stats<<<CH, 256, 0, stream>>>(cvt, mb, ib);
    bn_apply<<<(CH * SP) / 256, 256, 0, stream>>>(cvt, mb, ib, g2, b2, h2, 0);

    gram_gemm<<<dim3(50, 50), 256, 0, stream>>>(h2, out);
    softmax_rows<<<SP, 256, 0, stream>>>(out);
}
